// SNN2_63471026700413
// MI455X (gfx1250) — compile-verified, matrix operand order checked by reading, not run on hardware
//
#include <hip/hip_runtime.h>

// SNN fused forward: T=500 steps, B=4096, I=96, H=128, O=2
//   cur1 = x_t @ W1^T            -> v_wmma_f32_16x16x32_bf16 (f32 accum)
//   mem1 = (1-reset)*(tau*mem1+cur1), spk = mem1>1
//   mem2 = tau*mem2 + spk @ Wout^T ; out[t] = mem2
//
// One block = 16 batch rows (256 blocks for grid parallelism), 4 wave32 waves,
// each wave owns 2 N-tiles (cols n0..n0+31). mem1 lives in WMMA accum regs,
// W1 bf16 B-fragments are register-resident for all 500 steps. Spike tile is
// double-buffered in LDS -> exactly ONE barrier per timestep.

#define T_STEPS 500
#define BATCH   4096
#define IN_F    96
#define HID     128
#define OUT_F   2
#define TAU_F   0.96f
#define THR_F   1.0f
#define BT      16          // batch rows per block

typedef __attribute__((ext_vector_type(16))) __bf16 v16bf;
typedef __attribute__((ext_vector_type(8)))  float  v8f;

// Build one 16x32 bf16 fragment for this lane from a row-major f32 row.
// 16-bit A layout (ISA 7.12.2): lane half h, VGPR v (elem pair e):
//   K(v,h,e) = (v<4 ? 0 : 16) + 8*h + 2*(v%4) + e
// => elements 0..7  <- row[kbase + 8*h + 0..7]
//    elements 8..15 <- row[kbase + 16 + 8*h + 0..7]
// B (32x16, K x N) mirrors it with N = lane%16; column n of W1^T = row n of W1.
__device__ __forceinline__ v16bf load_frag_bf16(const float* __restrict__ row,
                                                int kbase, int h) {
  const float4* p0 = reinterpret_cast<const float4*>(row + kbase + 8 * h);
  const float4* p1 = reinterpret_cast<const float4*>(row + kbase + 16 + 8 * h);
  float4 r0 = p0[0], r1 = p0[1];
  float4 r2 = p1[0], r3 = p1[1];
  v16bf f;
  f[0]  = (__bf16)r0.x; f[1]  = (__bf16)r0.y; f[2]  = (__bf16)r0.z; f[3]  = (__bf16)r0.w;
  f[4]  = (__bf16)r1.x; f[5]  = (__bf16)r1.y; f[6]  = (__bf16)r1.z; f[7]  = (__bf16)r1.w;
  f[8]  = (__bf16)r2.x; f[9]  = (__bf16)r2.y; f[10] = (__bf16)r2.z; f[11] = (__bf16)r2.w;
  f[12] = (__bf16)r3.x; f[13] = (__bf16)r3.y; f[14] = (__bf16)r3.z; f[15] = (__bf16)r3.w;
  return f;
}

extern "C" __global__ __launch_bounds__(128)
void snn2_fused_kernel(const float* __restrict__ x,
                       const float* __restrict__ W1,
                       const float* __restrict__ Wout,
                       float* __restrict__ out) {
  __shared__ float spkLds[2][BT * HID];    // 2 x 8 KB, double-buffered spikes
  __shared__ float woutLds[OUT_F * HID];   // 1 KB

  const int tid  = threadIdx.x;
  const int lane = tid & 31;
  const int wave = tid >> 5;
  const int h    = lane >> 4;      // lane half (K-subset selector)
  const int lm   = lane & 15;      // A: row-in-tile, B/C: col-in-tile
  const int n0   = wave * 32;      // this wave's column base (2 N-tiles)
  const int blockRow = blockIdx.x * BT;

  // Stage Wout once.
  woutLds[tid]       = Wout[tid];
  woutLds[tid + 128] = Wout[tid + 128];

  // Register-resident W1 fragments: 2 N-tiles x 3 K-chunks (48 VGPRs/lane).
  v16bf bfrag[2][3];
#pragma unroll
  for (int nt = 0; nt < 2; ++nt) {
    const float* wrow = W1 + (size_t)(n0 + nt * 16 + lm) * IN_F;
#pragma unroll
    for (int kt = 0; kt < 3; ++kt)
      bfrag[nt][kt] = load_frag_bf16(wrow, kt * 32, h);
  }

  // mem1 state in C-layout regs: elem i -> row = i + 8*h, col = n0 + 16*nt + lm
  v8f mem1[2];
#pragma unroll
  for (int nt = 0; nt < 2; ++nt)
#pragma unroll
    for (int i = 0; i < 8; ++i) mem1[nt][i] = 0.0f;

  // Readout: 32 (row,o) pairs spread over all 4 waves (8 threads per wave).
  const bool ro_active = (lane < 8);
  const int  pairIdx   = wave * 8 + lane;   // 0..31 when active
  const int  ro_row    = pairIdx >> 1;
  const int  ro_o      = pairIdx & 1;
  float mem2 = 0.0f;

  const float* xrow = x + (size_t)(blockRow + lm) * IN_F;
  const size_t xstep = (size_t)BATCH * IN_F;

  __syncthreads();

  for (int t = 0; t < T_STEPS; ++t) {
    const float* xr = xrow + (size_t)t * xstep;
    if (t + 1 < T_STEPS) {   // gfx1250 global_prefetch_b8: next row = 3x128B lines
      __builtin_prefetch(xr + xstep, 0, 1);
      __builtin_prefetch(xr + xstep + 32, 0, 1);
      __builtin_prefetch(xr + xstep + 64, 0, 1);
    }

    // C = tau * mem1 ; then accumulate cur1 via WMMA over K=96.
    v8f c[2];
#pragma unroll
    for (int nt = 0; nt < 2; ++nt)
#pragma unroll
      for (int i = 0; i < 8; ++i) c[nt][i] = TAU_F * mem1[nt][i];

#pragma unroll
    for (int kt = 0; kt < 3; ++kt) {
      v16bf a = load_frag_bf16(xr, kt * 32, h);
#pragma unroll
      for (int nt = 0; nt < 2; ++nt)
        c[nt] = __builtin_amdgcn_wmma_f32_16x16x32_bf16(
            false, a, false, bfrag[nt][kt], (short)0, c[nt], false, false);
    }

    // Reset (from PREVIOUS membrane), spike, stash spikes to buffer t&1.
    float* sbuf = &spkLds[t & 1][0];
#pragma unroll
    for (int nt = 0; nt < 2; ++nt) {
      const int col = n0 + nt * 16 + lm;
#pragma unroll
      for (int i = 0; i < 8; ++i) {
        float base  = c[nt][i];                       // tau*mem1 + cur1
        float nmem  = (mem1[nt][i] > THR_F) ? 0.0f : base;
        mem1[nt][i] = nmem;
        sbuf[(i + 8 * h) * HID + col] = (nmem > THR_F) ? 1.0f : 0.0f;
      }
    }

    __syncthreads();   // the ONLY barrier per step (spike buffer is 2-deep)

    // Readout of step t overlaps step t+1's GEMM in non-readout lanes.
    if (ro_active) {
      const float4* sp = reinterpret_cast<const float4*>(&sbuf[ro_row * HID]);
      const float4* wo = reinterpret_cast<const float4*>(&woutLds[ro_o * HID]);
      float acc = 0.0f;
#pragma unroll
      for (int i = 0; i < 32; ++i) {
        float4 s = sp[i];
        float4 w = wo[i];
        acc += s.x * w.x + s.y * w.y + s.z * w.z + s.w * w.w;
      }
      mem2 = TAU_F * mem2 + acc;
      // Streaming output: non-temporal so 16MB of preds doesn't pollute L2.
      __builtin_nontemporal_store(
          mem2, &out[((size_t)t * BATCH + blockRow + ro_row) * OUT_F + ro_o]);
    }
    // No second barrier needed: step t+2 writes spkLds[t&1] only after
    // barrier t+1, which every readout thread reaches after finishing here.
  }
}

extern "C" void kernel_launch(void* const* d_in, const int* in_sizes, int n_in,
                              void* d_out, int out_size, void* d_ws, size_t ws_size,
                              hipStream_t stream) {
  const float* x    = (const float*)d_in[0];   // [500, 4096, 96]
  const float* W1   = (const float*)d_in[1];   // [128, 96]
  const float* Wout = (const float*)d_in[2];   // [2, 128]
  float* out = (float*)d_out;                  // [500, 4096, 2]
  (void)in_sizes; (void)n_in; (void)out_size; (void)d_ws; (void)ws_size;

  dim3 grid(BATCH / BT);   // 256 blocks
  dim3 block(128);         // 4 wave32 waves
  hipLaunchKernelGGL(snn2_fused_kernel, grid, block, 0, stream, x, W1, Wout, out);
}
